// EncoderDecoderAttention_33492154974536
// MI455X (gfx1250) — matrix operand order repaired and checked by reading, not compiled, not run on hardware
//
#include <hip/hip_runtime.h>
#include <hip/hip_bf16.h>

typedef __attribute__((ext_vector_type(16))) _Float16 v16h;
typedef __attribute__((ext_vector_type(8)))  _Float16 v8h;
typedef __attribute__((ext_vector_type(4)))  _Float16 v4h;
typedef __attribute__((ext_vector_type(8)))  float    v8f;

#define D_  1024
#define H_  16
#define HD_ 64
#define B_  2
#define T_  2048
#define M_  (B_ * T_)   // 4096 rows for all big GEMMs
#define K_  1024        // contraction dim for all big GEMMs

// A-fragment (16x32 f16) & B-fragment (32x16 f16): 8 VGPRs = 16 halves per lane.
union Frag16 { v16h v; v8h h[2]; };

__device__ __forceinline__ v8f wmma_f16(const Frag16& a, const Frag16& b, v8f c) {
  // D = A(16x32) * B(32x16) + C, f32 accumulate
  return __builtin_amdgcn_wmma_f32_16x16x32_f16(false, a.v, false, b.v,
                                                (short)0, c, false, false);
}

// ---------------------------------------------------------------- cvt f32->f16
__global__ void cvt_f32_to_f16(const float* __restrict__ in,
                               _Float16* __restrict__ out, int n) {
  int i = (blockIdx.x * blockDim.x + threadIdx.x) * 4;
  if (i + 3 < n) {
    float4 f = *(const float4*)(in + i);
    v4h o; o[0] = (_Float16)f.x; o[1] = (_Float16)f.y;
           o[2] = (_Float16)f.z; o[3] = (_Float16)f.w;
    *(v4h*)(out + i) = o;
  }
}

// ---------------- W[K][N] f32 -> Wt[N][K] f16 (B^T), LDS-tiled, coalesced
// grid (K/32, N/32), block 256 (32x8)
__global__ void wt_cvt_transpose(const float* __restrict__ W,
                                 _Float16* __restrict__ Wt, int K, int N) {
  __shared__ float tile[32][33];
  const int kb = blockIdx.x * 32;
  const int nb = blockIdx.y * 32;
  const int tx = threadIdx.x & 31;
  const int ty = threadIdx.x >> 5;
#pragma unroll
  for (int i = ty; i < 32; i += 8)
    tile[i][tx] = W[(size_t)(kb + i) * N + nb + tx];   // read rows of W, coalesced
  __syncthreads();
#pragma unroll
  for (int i = ty; i < 32; i += 8)
    Wt[(size_t)(nb + i) * K + kb + tx] = (_Float16)tile[tx][i];  // write rows of Wt
}

// ---------------------------------------------------------------- WMMA GEMM
// A[M][K] f16 row-major, Wt[N][K] f16 (i.e. B^T). Each wave: 32(M) x 64(N),
// so every B fragment is reused by two WMMAs (12 b128 loads per 8 WMMAs).
// All six fragments of a k-step are loaded into distinct registers before the
// WMMA chain so the compiler can clause the loads and use partial loadcnt waits.
// mode 0: f16 out [M][1024]   (Q, K projections)
// mode 1: f16 out transposed per head -> Vt[B][H*HD][T]
// mode 2: f32 out [M][1024]   (final projection -> d_out)
__global__ void gemm16(const _Float16* __restrict__ A,
                       const _Float16* __restrict__ Wt,
                       void* __restrict__ outp, int mode) {
  const int lane = threadIdx.x & 31;
  const int w    = threadIdx.x >> 5;
  const int hi   = lane >> 4;
  const int ln   = lane & 15;
  const int m0   = blockIdx.y * 256 + w * 32;   // wave covers rows m0..m0+31
  const int n0   = blockIdx.x * 64;

  v8f c[2][4] = {};
  for (int k0 = 0; k0 < K_; k0 += 32) {
    const _Float16* ar0 = A + (size_t)(m0 + ln) * K_ + k0;
    const _Float16* ar1 = ar0 + (size_t)16 * K_;
    const _Float16* br0 = Wt + (size_t)(n0 + ln) * K_ + k0 + 16 * hi;
    const _Float16* br1 = br0 + (size_t)16 * K_;
    const _Float16* br2 = br0 + (size_t)32 * K_;
    const _Float16* br3 = br0 + (size_t)48 * K_;

    Frag16 a0, a1, b0, b1, b2, b3;
    a0.h[0] = *(const v8h*)(ar0 + 16 * hi);       // v0..3 : k = 16*hi + 2v
    a0.h[1] = *(const v8h*)(ar0 + 16 * hi + 8);   // v4..7 : k = 16*hi + 2v
    a1.h[0] = *(const v8h*)(ar1 + 16 * hi);
    a1.h[1] = *(const v8h*)(ar1 + 16 * hi + 8);
    b0.h[0] = *(const v8h*)br0;  b0.h[1] = *(const v8h*)(br0 + 8);  // k = 16*hi+2v
    b1.h[0] = *(const v8h*)br1;  b1.h[1] = *(const v8h*)(br1 + 8);
    b2.h[0] = *(const v8h*)br2;  b2.h[1] = *(const v8h*)(br2 + 8);
    b3.h[0] = *(const v8h*)br3;  b3.h[1] = *(const v8h*)(br3 + 8);

    c[0][0] = wmma_f16(a0, b0, c[0][0]);
    c[1][0] = wmma_f16(a1, b0, c[1][0]);
    c[0][1] = wmma_f16(a0, b1, c[0][1]);
    c[1][1] = wmma_f16(a1, b1, c[1][1]);
    c[0][2] = wmma_f16(a0, b2, c[0][2]);
    c[1][2] = wmma_f16(a1, b2, c[1][2]);
    c[0][3] = wmma_f16(a0, b3, c[0][3]);
    c[1][3] = wmma_f16(a1, b3, c[1][3]);
  }

  if (mode == 0) {
    _Float16* out = (_Float16*)outp;
#pragma unroll
    for (int mt = 0; mt < 2; ++mt)
#pragma unroll
      for (int nc = 0; nc < 4; ++nc)
#pragma unroll
        for (int v = 0; v < 8; ++v)
          out[(size_t)(m0 + mt * 16 + v + 8 * hi) * 1024 + n0 + nc * 16 + ln] =
              (_Float16)c[mt][nc][v];
  } else if (mode == 1) {
    _Float16* Vt = (_Float16*)outp;
    int bb = m0 >> 11;                 // m0 / T_  (32-row tile never straddles b)
#pragma unroll
    for (int mt = 0; mt < 2; ++mt) {
      int t0 = (m0 + mt * 16) & (T_ - 1);
#pragma unroll
      for (int nc = 0; nc < 4; ++nc) {
        int f = n0 + nc * 16 + ln;               // feature = h*64 + d
        v8h pk;
#pragma unroll
        for (int v = 0; v < 8; ++v) pk[v] = (_Float16)c[mt][nc][v];
        *(v8h*)(Vt + ((size_t)(bb * (H_ * HD_) + f)) * T_ + t0 + 8 * hi) = pk;
      }
    }
  } else {
    float* out = (float*)outp;
#pragma unroll
    for (int mt = 0; mt < 2; ++mt)
#pragma unroll
      for (int nc = 0; nc < 4; ++nc)
#pragma unroll
        for (int v = 0; v < 8; ++v)
          out[(size_t)(m0 + mt * 16 + v + 8 * hi) * 1024 + n0 + nc * 16 + ln] =
              c[mt][nc][v];
  }
}

// ------------------------------------------------------ flash attention (f16)
// grid (T/128, H, B), block 256 = 8 waves; each wave owns a 16-row q tile.
__global__ void flash_attn(const _Float16* __restrict__ Q,
                           const _Float16* __restrict__ Kh,
                           const _Float16* __restrict__ Vt,
                           _Float16* __restrict__ Y) {
  __shared__ _Float16 plds[8 * 512];            // 16x32 P tile per wave
  const int lane = threadIdx.x & 31;
  const int w    = threadIdx.x >> 5;
  const int hi   = lane >> 4;
  const int ln   = lane & 15;
  const int b    = blockIdx.z, h = blockIdx.y;
  const int q0   = blockIdx.x * 128 + w * 16;

  // Q tile as two A-fragments (k-dims 0..31 and 32..63), held in registers.
  Frag16 qa0, qa1;
  const _Float16* qr = Q + (size_t)(b * T_ + q0 + ln) * D_ + h * HD_;
  qa0.h[0] = *(const v8h*)(qr + 16 * hi);
  qa0.h[1] = *(const v8h*)(qr + 8 + 16 * hi);
  qa1.h[0] = *(const v8h*)(qr + 32 + 16 * hi);
  qa1.h[1] = *(const v8h*)(qr + 40 + 16 * hi);

  float mrow[8], lrow[8];
#pragma unroll
  for (int v = 0; v < 8; ++v) { mrow[v] = -3.0e38f; lrow[v] = 0.f; }
  v8f o[4] = {};

  _Float16* pl = plds + w * 512;
  const _Float16* kbase = Kh + (size_t)(b * T_) * D_ + h * HD_;
  const _Float16* vbase = Vt + (size_t)(b * (H_ * HD_) + h * HD_) * T_;

  for (int kt = 0; kt < T_; kt += 32) {
    if (kt + 32 < T_)
      __builtin_prefetch(kbase + (size_t)(kt + 32 + ln) * D_, 0, 1);

    // S = (Q . K^T) for 32 keys -> two 16x16 f32 tiles.
    // Load all four K fragments first, then run the WMMA chain.
    v8f s0 = {}, s1 = {};
    {
      const _Float16* kr0 = kbase + (size_t)(kt + ln) * D_ + 16 * hi;
      const _Float16* kr1 = kbase + (size_t)(kt + 16 + ln) * D_ + 16 * hi;
      Frag16 b0, b1, c0, c1;
      b0.h[0] = *(const v8h*)kr0;         b0.h[1] = *(const v8h*)(kr0 + 8);
      b1.h[0] = *(const v8h*)(kr0 + 32);  b1.h[1] = *(const v8h*)(kr0 + 40);
      c0.h[0] = *(const v8h*)kr1;         c0.h[1] = *(const v8h*)(kr1 + 8);
      c1.h[0] = *(const v8h*)(kr1 + 32);  c1.h[1] = *(const v8h*)(kr1 + 40);
      s0 = wmma_f16(qa0, b0, s0);
      s1 = wmma_f16(qa0, c0, s1);
      s0 = wmma_f16(qa1, b1, s0);
      s1 = wmma_f16(qa1, c1, s1);
    }

    // online softmax over this 32-key chunk (rows live in 16-lane halves)
    const float sc = 0.125f;                    // 1/sqrt(64)
    float cmax[8];
#pragma unroll
    for (int v = 0; v < 8; ++v) {
      s0[v] *= sc; s1[v] *= sc;
      cmax[v] = fmaxf(s0[v], s1[v]);
    }
#pragma unroll
    for (int off = 1; off < 16; off <<= 1)
#pragma unroll
      for (int v = 0; v < 8; ++v)
        cmax[v] = fmaxf(cmax[v], __shfl_xor(cmax[v], off, 32));

    float rsum[8];
#pragma unroll
    for (int v = 0; v < 8; ++v) {
      float mnew = fmaxf(mrow[v], cmax[v]);
      float corr = __expf(mrow[v] - mnew);
      mrow[v] = mnew;
      lrow[v] *= corr;
      float p0 = __expf(s0[v] - mnew);
      float p1 = __expf(s1[v] - mnew);
      s0[v] = p0; s1[v] = p1;
      rsum[v] = p0 + p1;
      o[0][v] *= corr; o[1][v] *= corr; o[2][v] *= corr; o[3][v] *= corr;
    }
#pragma unroll
    for (int off = 1; off < 16; off <<= 1)
#pragma unroll
      for (int v = 0; v < 8; ++v)
        rsum[v] += __shfl_xor(rsum[v], off, 32);
#pragma unroll
    for (int v = 0; v < 8; ++v) lrow[v] += rsum[v];

    // P (C layout) -> LDS -> A fragment layout
#pragma unroll
    for (int v = 0; v < 8; ++v) {
      int r = v + 8 * hi;
      pl[r * 32 + ln]      = (_Float16)s0[v];
      pl[r * 32 + 16 + ln] = (_Float16)s1[v];
    }
    asm volatile("s_wait_dscnt 0" ::: "memory");
    Frag16 pf;
    const _Float16* pr = pl + ln * 32;
    pf.h[0] = *(const v8h*)(pr + 16 * hi);
    pf.h[1] = *(const v8h*)(pr + 8 + 16 * hi);

    // O += P(16x32) . V(32x64), V read from transposed layout (contig keys).
    // Load all four V fragments before the WMMA chain.
    {
      const _Float16* vr0 = vbase + (size_t)(ln) * T_ + kt + 16 * hi;
      const _Float16* vr1 = vr0 + (size_t)16 * T_;
      const _Float16* vr2 = vr0 + (size_t)32 * T_;
      const _Float16* vr3 = vr0 + (size_t)48 * T_;
      Frag16 v0, v1, v2, v3;
      v0.h[0] = *(const v8h*)vr0; v0.h[1] = *(const v8h*)(vr0 + 8);
      v1.h[0] = *(const v8h*)vr1; v1.h[1] = *(const v8h*)(vr1 + 8);
      v2.h[0] = *(const v8h*)vr2; v2.h[1] = *(const v8h*)(vr2 + 8);
      v3.h[0] = *(const v8h*)vr3; v3.h[1] = *(const v8h*)(vr3 + 8);
      o[0] = wmma_f16(pf, v0, o[0]);
      o[1] = wmma_f16(pf, v1, o[1]);
      o[2] = wmma_f16(pf, v2, o[2]);
      o[3] = wmma_f16(pf, v3, o[3]);
    }
  }

  // normalize and store Y (f16, natural [B,T,D])
#pragma unroll
  for (int v = 0; v < 8; ++v) {
    float inv = 1.0f / lrow[v];
    o[0][v] *= inv; o[1][v] *= inv; o[2][v] *= inv; o[3][v] *= inv;
  }
  _Float16* yb = Y + (size_t)(b * T_ + q0) * D_ + h * HD_;
#pragma unroll
  for (int v = 0; v < 8; ++v) {
    int r = v + 8 * hi;
    yb[(size_t)r * D_ + ln]      = (_Float16)o[0][v];
    yb[(size_t)r * D_ + 16 + ln] = (_Float16)o[1][v];
    yb[(size_t)r * D_ + 32 + ln] = (_Float16)o[2][v];
    yb[(size_t)r * D_ + 48 + ln] = (_Float16)o[3][v];
  }
}

// ---------------------------------------------------------------- launcher
extern "C" void kernel_launch(void* const* d_in, const int* in_sizes, int n_in,
                              void* d_out, int out_size, void* d_ws, size_t ws_size,
                              hipStream_t stream) {
  const float* x   = (const float*)d_in[0];
  const float* e   = (const float*)d_in[1];
  const float* Wen = (const float*)d_in[2];   // [1024, 2048]
  const float* Wq  = (const float*)d_in[3];   // [1024, 1024]
  const float* Wo  = (const float*)d_in[4];   // [1024, 1024]

  char* ws = (char*)d_ws;
  _Float16* Xh  = (_Float16*)ws; ws += (size_t)M_ * K_ * 2;
  _Float16* Eh  = (_Float16*)ws; ws += (size_t)M_ * K_ * 2;
  _Float16* Wtq = (_Float16*)ws; ws += (size_t)1024 * K_ * 2;
  _Float16* Wte = (_Float16*)ws; ws += (size_t)2048 * K_ * 2;
  _Float16* Wto = (_Float16*)ws; ws += (size_t)1024 * K_ * 2;
  _Float16* Qh  = (_Float16*)ws; ws += (size_t)M_ * 1024 * 2;
  _Float16* Kh  = (_Float16*)ws; ws += (size_t)M_ * 1024 * 2;
  _Float16* Vt  = (_Float16*)ws; ws += (size_t)M_ * 1024 * 2;
  _Float16* Yh  = (_Float16*)ws; ws += (size_t)M_ * 1024 * 2;

  const int nX = M_ * K_;
  cvt_f32_to_f16<<<nX / (256 * 4), 256, 0, stream>>>(x, Xh, nX);
  cvt_f32_to_f16<<<nX / (256 * 4), 256, 0, stream>>>(e, Eh, nX);
  wt_cvt_transpose<<<dim3(K_ / 32, 1024 / 32), 256, 0, stream>>>(Wq,  Wtq, K_, 1024);
  wt_cvt_transpose<<<dim3(K_ / 32, 2048 / 32), 256, 0, stream>>>(Wen, Wte, K_, 2048);
  wt_cvt_transpose<<<dim3(K_ / 32, 1024 / 32), 256, 0, stream>>>(Wo,  Wto, K_, 1024);

  dim3 gg(1024 / 64, M_ / 256);
  gemm16<<<gg, 256, 0, stream>>>(Xh, Wtq, (void*)Qh, 0);                      // Q
  gemm16<<<gg, 256, 0, stream>>>(Eh, Wte, (void*)Kh, 0);                      // K
  gemm16<<<gg, 256, 0, stream>>>(Eh, Wte + (size_t)1024 * K_, (void*)Vt, 1);  // V^T

  dim3 ga(T_ / 128, H_, B_);
  flash_attn<<<ga, 256, 0, stream>>>(Qh, Kh, Vt, Yh);

  gemm16<<<gg, 256, 0, stream>>>(Yh, Wto, d_out, 2);                          // y @ W_o
}
